// SlayerLoihiMLP_27857157882557
// MI455X (gfx1250) — matrix operand order repaired and checked by reading, not compiled, not run on hardware
//
#include <hip/hip_runtime.h>
#include <hip/hip_bf16.h>

typedef __attribute__((ext_vector_type(16))) _Float16 v16h;
typedef __attribute__((ext_vector_type(8)))  _Float16 v8h;
typedef __attribute__((ext_vector_type(8)))  float    v8f;

#define AS1 __attribute__((address_space(1)))

#define T_STEPS 256
#define B_SZ    32
#define IN_SZ   1024
#define HID_SZ  2048
#define OUT_SZ  512
#define MTOT    (T_STEPS * B_SZ)   // 8192 rows in the time-batched GEMMs

// ---------------------------------------------------------------------------
// C[M,N] = A[M,K] * W[N,K]^T   (A,W fp16 row-major; C f32 row-major)
// Block = 8 waves, block tile 128x128, wave tile 32x64 (2x4 WMMA accumulators).
// Explicit addrspace(1) pointers keep all accesses on the GLOBAL path
// (LOADcnt/STOREcnt only — no FLAT/DScnt coupling), with per-fragment base
// pointers advancing by a constant so loads use immediate offsets.
//
// Fragment layouts used here (wave32, lane = r + 16*s; contiguous K, same order for A and B):
//   A 16x32 f16 : lane holds row r, K = 16s..16s+15 contiguous (two b128 loads)
//   B 32x16 f16 : lane holds col r, K = 16s..16s+15 contiguous (one v16h load)
//   C/D 16x16 f32: lane holds col r, rows 8s+j in VGPR j
// ---------------------------------------------------------------------------
__global__ __launch_bounds__(256)
void wmma_gemm_nt(const _Float16* __restrict__ A_, const _Float16* __restrict__ W_,
                  float* __restrict__ C_, int M, int N, int K) {
  const AS1 _Float16* A = (const AS1 _Float16*)A_;
  const AS1 _Float16* W = (const AS1 _Float16*)W_;
  AS1 float*          C = (AS1 float*)C_;

  const int lane = threadIdx.x & 31;
  const int wave = threadIdx.x >> 5;   // 0..7
  const int wm = wave >> 1;            // 0..3 -> 128 M per block
  const int wn = wave & 1;             // 0..1 -> 128 N per block
  const int m0 = blockIdx.x * 128 + wm * 32;
  const int n0 = blockIdx.y * 128 + wn * 64;
  const int s = lane >> 4;             // half-wave select
  const int r = lane & 15;

  // Per-fragment base pointers (lane-dependent part folded in once).
  const AS1 _Float16* pa[2];
  const AS1 _Float16* pb[4];
#pragma unroll
  for (int i = 0; i < 2; ++i)
    pa[i] = A + (size_t)(m0 + 16 * i + r) * (size_t)K + 16 * s;
#pragma unroll
  for (int j = 0; j < 4; ++j)
    pb[j] = W + (size_t)(n0 + 16 * j + r) * (size_t)K + 16 * s;

  v8f acc[2][4] = {};

#pragma unroll 2
  for (int k = 0; k < K; k += 32) {
    v16h af[2], bf[4];
#pragma unroll
    for (int i = 0; i < 2; ++i) {            // A: two b128 at +0 / +16 bytes
      v8h lo = *(const AS1 v8h*)(pa[i]);
      v8h hi = *(const AS1 v8h*)(pa[i] + 8);
#pragma unroll
      for (int e = 0; e < 8; ++e) { af[i][e] = lo[e]; af[i][e + 8] = hi[e]; }
      pa[i] += 32;
    }
#pragma unroll
    for (int j = 0; j < 4; ++j) {            // B: 32 contiguous bytes
      bf[j] = *(const AS1 v16h*)(pb[j]);
      pb[j] += 32;
    }
#pragma unroll
    for (int i = 0; i < 2; ++i)
#pragma unroll
      for (int j = 0; j < 4; ++j)
        acc[i][j] = __builtin_amdgcn_wmma_f32_16x16x32_f16(
            false, af[i], false, bf[j], (short)0, acc[i][j], false, false);
  }

  const unsigned uN = (unsigned)N;
#pragma unroll
  for (int i = 0; i < 2; ++i)
#pragma unroll
    for (int j = 0; j < 4; ++j) {
      const unsigned base =
          (unsigned)(m0 + 16 * i + 8 * s) * uN + (unsigned)(n0 + 16 * j + r);
#pragma unroll
      for (int jj = 0; jj < 8; ++jj)
        C[base + (unsigned)jj * uN] = acc[i][j][jj];   // coalesced across lanes
    }
}

// ---------------------------------------------------------------------------
// quantize_w (round-to-even multiples of 2) + convert to fp16 (exact: even ints)
// ---------------------------------------------------------------------------
__global__ void quant_to_h(const float* __restrict__ w, _Float16* __restrict__ o, int n) {
  int i = blockIdx.x * 256 + threadIdx.x;
  if (i < n) o[i] = (_Float16)(rintf(w[i] * 0.5f) * 2.0f);  // RNE like jnp.round
}

// ---------------------------------------------------------------------------
// spikes [B, IN, T] f32 -> Xh [T, B, IN] fp16, LDS-tiled transpose (32x32)
// ---------------------------------------------------------------------------
__global__ __launch_bounds__(256)
void transpose_spikes(const float* __restrict__ sp, _Float16* __restrict__ xh) {
  __shared__ float tile[32][33];
  const int b  = blockIdx.z;
  const int i0 = blockIdx.x * 32;
  const int t0 = blockIdx.y * 32;
  const int tx = threadIdx.x;           // 0..31
  for (int r = threadIdx.y; r < 32; r += 8)   // coalesced along t
    tile[r][tx] = sp[((size_t)b * IN_SZ + (i0 + r)) * T_STEPS + t0 + tx];
  __syncthreads();
  for (int r = threadIdx.y; r < 32; r += 8)   // coalesced along i
    xh[((size_t)(t0 + r) * B_SZ + b) * IN_SZ + i0 + tx] = (_Float16)tile[tx][r];
}

// ---------------------------------------------------------------------------
// Loihi CUBA neuron scan. REF_DELAY=1 => ref is identically 0 (dead branch).
// I = floor(0.75 I) + 64 x ; V = floor(0.96875 V) + I ; spike at V>=5120, reset.
// Output is the delay-shifted spike train in fp16 [T, BC] (feeds next GEMM).
// ---------------------------------------------------------------------------
__global__ void neuron_scan_h(const float* __restrict__ inp,
                              _Float16* __restrict__ s_del, int BC) {
  const int idx = blockIdx.x * 256 + threadIdx.x;
  if (idx >= BC) return;
  float I = 0.f, V = 0.f;
  s_del[idx] = (_Float16)0.f;                       // t = 0 (delay shift)
  for (int t = 0; t < T_STEPS; ++t) {
    float x = inp[(size_t)t * BC + idx];
    I = floorf(I * 0.75f) + 64.0f * x;
    V = floorf(V * 0.96875f) + I;
    float s = (V >= 5120.0f) ? 1.0f : 0.0f;
    V = V * (1.0f - s);
    if (t + 1 < T_STEPS) s_del[(size_t)(t + 1) * BC + idx] = (_Float16)s;
  }
}

// Final layer: scan + delay shift + write f32 output transposed to [B, OUT, T]
__global__ void neuron_scan_out(const float* __restrict__ inp,
                                float* __restrict__ out) {
  const int idx = blockIdx.x * 256 + threadIdx.x;   // over B*OUT
  const int BC = B_SZ * OUT_SZ;
  if (idx >= BC) return;
  float* o = out + (size_t)idx * T_STEPS;           // out[b][c][:] contiguous
  float I = 0.f, V = 0.f;
  o[0] = 0.0f;
  for (int t = 0; t < T_STEPS; ++t) {
    float x = inp[(size_t)t * BC + idx];
    I = floorf(I * 0.75f) + 64.0f * x;
    V = floorf(V * 0.96875f) + I;
    float s = (V >= 5120.0f) ? 1.0f : 0.0f;
    V = V * (1.0f - s);
    if (t + 1 < T_STEPS) o[t + 1] = s;
  }
}

// ---------------------------------------------------------------------------
extern "C" void kernel_launch(void* const* d_in, const int* in_sizes, int n_in,
                              void* d_out, int out_size, void* d_ws, size_t ws_size,
                              hipStream_t stream) {
  const float* sp = (const float*)d_in[0];   // [B, IN, T]
  const float* W1 = (const float*)d_in[1];   // [HID, IN]
  const float* W2 = (const float*)d_in[2];   // [OUT, HID]
  float* out = (float*)d_out;                // [B, OUT, T]
  char* ws = (char*)d_ws;

  // workspace layout (peak ~118 MB; O aliases H which is dead by then)
  _Float16* Xh  = (_Float16*)(ws + 0);          // 16 MB  [T*B, IN] fp16
  _Float16* W1h = (_Float16*)(ws + 16777216);   //  4 MB  [HID, IN] fp16
  _Float16* W2h = (_Float16*)(ws + 20971520);   //  2 MB  [OUT, HID] fp16
  _Float16* S1h = (_Float16*)(ws + 23068672);   // 32 MB  [T*B, HID] fp16
  float*    H   = (float*)   (ws + 56623104);   // 64 MB  [T*B, HID] f32
  float*    O   = (float*)   (ws + 56623104);   // 16 MB  [T*B, OUT] f32 (reuses H)

  // 1) weight quantize + fp16 convert
  quant_to_h<<<(HID_SZ * IN_SZ + 255) / 256, 256, 0, stream>>>(W1, W1h, HID_SZ * IN_SZ);
  quant_to_h<<<(OUT_SZ * HID_SZ + 255) / 256, 256, 0, stream>>>(W2, W2h, OUT_SZ * HID_SZ);

  // 2) spike transpose [B,IN,T] -> [T,B,IN] fp16
  {
    dim3 blk(32, 8);
    dim3 grd(IN_SZ / 32, T_STEPS / 32, B_SZ);
    transpose_spikes<<<grd, blk, 0, stream>>>(sp, Xh);
  }

  // 3) FC1: H[8192,2048] = Xh[8192,1024] x W1h[2048,1024]^T
  {
    dim3 grd(MTOT / 128, HID_SZ / 128);  // 64 x 16
    wmma_gemm_nt<<<grd, 256, 0, stream>>>(Xh, W1h, H, MTOT, HID_SZ, IN_SZ);
  }

  // 4) neuron scan layer 1 -> delayed fp16 spikes S1h
  neuron_scan_h<<<(B_SZ * HID_SZ) / 256, 256, 0, stream>>>(H, S1h, B_SZ * HID_SZ);

  // 5) FC2: O[8192,512] = S1h[8192,2048] x W2h[512,2048]^T
  {
    dim3 grd(MTOT / 128, OUT_SZ / 128);  // 64 x 4
    wmma_gemm_nt<<<grd, 256, 0, stream>>>(S1h, W2h, O, MTOT, OUT_SZ, HID_SZ);
  }

  // 6) neuron scan layer 2 + delay shift + transpose to [B, OUT, T]
  neuron_scan_out<<<(B_SZ * OUT_SZ) / 256, 256, 0, stream>>>(O, out);
}
